// MoE_32581621907468
// MI455X (gfx1250) — matrix operand order repaired and checked by reading, not compiled, not run on hardware
//
#include <hip/hip_runtime.h>
#include <hip/hip_bf16.h>

// MoE (top-2 of 8 experts) + shared expert, fp32 reference accuracy via
// bf16x3 split WMMA (AhBh + AhBl + AlBh, f32 accumulate) on gfx1250.
//
// Roofline: ~52 GFLOP compute; weights (~150MB) are L2-resident (192MB L2),
// HBM floor ~7us @ 23.3TB/s.
// This version
// pre-splits all operands ONCE (weights + activations) into bf16 hi/lo
// buffers (same total bytes as the f32 originals), so GEMM inner loops are
// pure b128 loads + v_wmma_f32_16x16x32_bf16.

#define Ntok 2048
#define Cdim 1024
#define Hdim 2048
#define Eexp 8
#define SCAP 4224          // 2*Ntok + 8*16 padding, 16-aligned per expert
#define RT   (SCAP / 16)   // 264 row tiles
#define EPSR 1.1920929e-07f

typedef __attribute__((ext_vector_type(16))) __bf16 v16bf;
typedef __attribute__((ext_vector_type(8)))  __bf16 v8bf;
typedef __attribute__((ext_vector_type(8)))  float  v8f;

__device__ __forceinline__ __bf16 f2bf(float f) {
    union { float f; unsigned u; } in; in.f = f;
    unsigned u = in.u;
    u += 0x7FFFu + ((u >> 16) & 1u);           // round-to-nearest-even
    union { unsigned short s; __bf16 b; } out; out.s = (unsigned short)(u >> 16);
    return out.b;
}
__device__ __forceinline__ float bf2f(__bf16 b) {
    union { unsigned short s; __bf16 b; } in; in.b = b;
    union { unsigned u; float f; } out; out.u = ((unsigned)in.s) << 16;
    return out.f;
}
__device__ __forceinline__ void splitTo(float f, __bf16& h, __bf16& l) {
    h = f2bf(f);
    l = f2bf(f - bf2f(h));   // f-hi exact (Sterbenz), lo error ~2^-17 relative
}

// A-matrix 16x32 bf16 layout (ISA 7.12.2): lanes 0-15 (row M=lane) hold
// K = kb+0..7 (elems 0-7) and kb+16..23 (elems 8-15); lanes 16-31 hold
// K = kb+8..15 and kb+24..31.  hoff = (lane>=16) ? 8 : 0.
__device__ __forceinline__ void loadA_pre(const __bf16* __restrict__ hrow,
                                          const __bf16* __restrict__ lrow,
                                          int kb, int hoff, v16bf& ah, v16bf& al) {
    union { v16bf v; v8bf h[2]; } U, V;
    U.h[0] = *(const v8bf*)(hrow + kb + hoff);
    U.h[1] = *(const v8bf*)(hrow + kb + 16 + hoff);
    V.h[0] = *(const v8bf*)(lrow + kb + hoff);
    V.h[1] = *(const v8bf*)(lrow + kb + 16 + hoff);
    ah = U.v; al = V.v;
}

// B-matrix 32x16 bf16 layout: lane column N = lane&15; lanes 0-15 hold
// K = kb+0..7 and kb+16..23; lanes 16-31 hold K = kb+8..15 and kb+24..31.
// hoff16 = (lane>=16) ? 16 : 0.
__device__ __forceinline__ void loadB_pre(const __bf16* __restrict__ hrow,
                                          const __bf16* __restrict__ lrow,
                                          int kb, int hoff16, v16bf& bh, v16bf& bl) {
    union { v16bf v; v8bf h[2]; } U, V;
    U.h[0] = *(const v8bf*)(hrow + kb + hoff16 / 2);
    U.h[1] = *(const v8bf*)(hrow + kb + 16 + hoff16 / 2);
    V.h[0] = *(const v8bf*)(lrow + kb + hoff16 / 2);
    V.h[1] = *(const v8bf*)(lrow + kb + 16 + hoff16 / 2);
    bh = U.v; bl = V.v;
}

#define WMMA(acc, A, B) \
    (acc) = __builtin_amdgcn_wmma_f32_16x16x32_bf16(false, (A), false, (B), (short)0, (acc), false, false)

// ---------------- Kernel 0: one-time f32 -> bf16 hi/lo split ----------------
__global__ __launch_bounds__(256) void k_split(
    const float* __restrict__ src, __bf16* __restrict__ hi,
    __bf16* __restrict__ lo, int nquad)   // nquad = elements/4
{
    int i = blockIdx.x * 256 + threadIdx.x;
    if (i >= nquad) return;
    float4 v = ((const float4*)src)[i];
    union { __bf16 e[4]; unsigned long long u; } H, L;
    splitTo(v.x, H.e[0], L.e[0]);
    splitTo(v.y, H.e[1], L.e[1]);
    splitTo(v.z, H.e[2], L.e[2]);
    splitTo(v.w, H.e[3], L.e[3]);
    ((unsigned long long*)hi)[i] = H.u;
    ((unsigned long long*)lo)[i] = L.u;
}

// ---------------- Kernel 1: rmsnorm + gating + shared gate ----------------
__global__ __launch_bounds__(256) void k_gating(
    const float* __restrict__ x, const float* __restrict__ gate_w,
    const float* __restrict__ sgw,
    __bf16* __restrict__ xthi, __bf16* __restrict__ xtlo,
    __bf16* __restrict__ xshi, __bf16* __restrict__ xslo,
    float* __restrict__ scores, int* __restrict__ tidx,
    float* __restrict__ tkw, float* __restrict__ sgv)
{
    __shared__ float sxt[Cdim];
    __shared__ float wred[8];
    __shared__ float raw[Eexp + 1];
    int n = blockIdx.x;
    int tid = threadIdx.x, lane = tid & 31, wv = tid >> 5;
    const float* xr = x + (size_t)n * Cdim;

    float s = 0.f;
    for (int i = tid; i < Cdim; i += 256) { float v = xr[i]; s += v * v; sxt[i] = v; }
    for (int o = 16; o > 0; o >>= 1) s += __shfl_down(s, o, 32);
    if (lane == 0) wred[wv] = s;
    __syncthreads();
    if (tid == 0) { float t = 0; for (int i = 0; i < 8; ++i) t += wred[i]; wred[0] = t; }
    __syncthreads();
    float msq = wred[0] * (1.0f / Cdim);
    float r  = rsqrtf(msq + EPSR);
    float r2 = rsqrtf(r * r * msq + EPSR);
    for (int i = tid; i < Cdim; i += 256) {
        float v = sxt[i] * r;
        sxt[i] = v;
        __bf16 h, l;
        splitTo(v, h, l);
        xthi[(size_t)n * Cdim + i] = h; xtlo[(size_t)n * Cdim + i] = l;
        splitTo(v * r2, h, l);
        xshi[(size_t)n * Cdim + i] = h; xslo[(size_t)n * Cdim + i] = l;
    }
    __syncthreads();
    {   // wave wv computes gate dot for expert wv (f32, from LDS)
        const float* gw = gate_w + (size_t)wv * Cdim;
        float d = 0.f;
        for (int i = lane; i < Cdim; i += 32) d += sxt[i] * gw[i];
        for (int o = 16; o > 0; o >>= 1) d += __shfl_down(d, o, 32);
        if (lane == 0) raw[wv] = d;
    }
    if (wv == 0) {  // shared gate dot
        float d = 0.f;
        for (int i = lane; i < Cdim; i += 32) d += sxt[i] * sgw[i];
        for (int o = 16; o > 0; o >>= 1) d += __shfl_down(d, o, 32);
        if (lane == 0) raw[8] = d;
    }
    __syncthreads();
    if (tid == 0) {
        float mx = -1e30f;
        for (int e = 0; e < 8; ++e) mx = fmaxf(mx, raw[e]);
        float sc[8], se = 0.f;
        for (int e = 0; e < 8; ++e) { sc[e] = expf(raw[e] - mx); se += sc[e]; }
        float inv = 1.f / se;
        int i0 = 0; float b0 = -1.f;
        for (int e = 0; e < 8; ++e) {
            sc[e] *= inv;
            scores[(size_t)n * 8 + e] = sc[e];
            if (sc[e] > b0) { b0 = sc[e]; i0 = e; }
        }
        int i1 = 0; float b1 = -1.f;
        for (int e = 0; e < 8; ++e)
            if (e != i0 && sc[e] > b1) { b1 = sc[e]; i1 = e; }
        float sw = 1.f / (b0 + b1 + 1e-6f);
        tidx[n * 2 + 0] = i0; tidx[n * 2 + 1] = i1;
        tkw[n * 2 + 0] = b0 * sw; tkw[n * 2 + 1] = b1 * sw;
        sgv[n] = 1.f / (1.f + expf(-raw[8]));
    }
}

// ---------------- Kernel 2: deterministic counting-sort scatter ----------------
__global__ __launch_bounds__(256) void k_scatter(
    const int* __restrict__ tidx, const float* __restrict__ tkw,
    int* __restrict__ cnt, int* __restrict__ off, int* __restrict__ perm,
    float* __restrict__ pw, int* __restrict__ t2s, int* __restrict__ texp)
{
    int tid = threadIdx.x;
    __shared__ int soff[Eexp + 1];
    if (tid < Eexp) {
        int c = 0;
        for (int t = 0; t < Ntok; ++t) {
            if (tidx[t * 2 + 0] == tid) ++c;
            if (tidx[t * 2 + 1] == tid) ++c;
        }
        cnt[tid] = c;
    }
    __syncthreads();
    if (tid == 0) {
        int o = 0;
        for (int e = 0; e < Eexp; ++e) { soff[e] = o; off[e] = o; o += (cnt[e] + 15) & ~15; }
        soff[Eexp] = o; off[Eexp] = o;
    }
    __syncthreads();
    for (int i = tid; i < SCAP; i += 256) { perm[i] = 0; pw[i] = 0.f; }
    __syncthreads();
    if (tid < Eexp) {
        int pos = soff[tid];
        for (int t = 0; t < Ntok; ++t)
            for (int k = 0; k < 2; ++k)
                if (tidx[t * 2 + k] == tid) {
                    perm[pos] = t; pw[pos] = tkw[t * 2 + k];
                    t2s[t * 2 + k] = pos; ++pos;
                }
    }
    __syncthreads();
    for (int tt = tid; tt < RT; tt += 256) {
        int sb = tt * 16, ee = 0;
        for (int q = 0; q < Eexp; ++q)
            if (sb >= soff[q] && sb < soff[q + 1]) ee = q;
        texp[tt] = ee;
    }
}

// ---------------- Kernel 3: grouped up-proj (xt @ w1[e]^T), relu^2 * gate ----------------
__global__ __launch_bounds__(256) void k_up(
    const __bf16* __restrict__ xthi, const __bf16* __restrict__ xtlo,
    const __bf16* __restrict__ w1hi, const __bf16* __restrict__ w1lo,
    const int* __restrict__ perm, const float* __restrict__ pw,
    const int* __restrict__ texp,
    __bf16* __restrict__ hhi, __bf16* __restrict__ hlo)
{
    int tile = blockIdx.x;
    int wv = threadIdx.x >> 5, lane = threadIdx.x & 31;
    int lrow = lane & 15, hi = lane >> 4;
    int sb = tile * 16;
    int e = texp[tile];
    int hcol = blockIdx.y * 128 + wv * 16 + lrow;
    int tok = perm[sb + lrow];
    const __bf16* arh = xthi + (size_t)tok * Cdim;
    const __bf16* arl = xtlo + (size_t)tok * Cdim;
    const __bf16* brh = w1hi + ((size_t)e * Hdim + hcol) * Cdim;
    const __bf16* brl = w1lo + ((size_t)e * Hdim + hcol) * Cdim;
    int hoff = hi ? 8 : 0, hoff16 = hi ? 16 : 0;
    v8f acc = {};
    for (int kb = 0; kb < Cdim; kb += 32) {
        v16bf ah, al, bh, bl;
        loadA_pre(arh, arl, kb, hoff, ah, al);
        loadB_pre(brh, brl, kb, hoff16, bh, bl);
        WMMA(acc, ah, bh); WMMA(acc, ah, bl); WMMA(acc, al, bh);
    }
#pragma unroll
    for (int r2 = 0; r2 < 8; ++r2) {
        int row = r2 + (hi ? 8 : 0);
        float g = pw[sb + row];
        float v = fmaxf(acc[r2], 0.f);
        v = v * v * g;
        __bf16 h, l; splitTo(v, h, l);
        size_t o = (size_t)(sb + row) * Hdim + hcol;
        hhi[o] = h; hlo[o] = l;
    }
}

// ---------------- Kernel 4: grouped down-proj (h @ w2[e]^T) -> slotout ----------------
__global__ __launch_bounds__(256) void k_down(
    const __bf16* __restrict__ hhi, const __bf16* __restrict__ hlo,
    const __bf16* __restrict__ w2hi, const __bf16* __restrict__ w2lo,
    const int* __restrict__ texp, float* __restrict__ slotout)
{
    int tile = blockIdx.x;
    int wv = threadIdx.x >> 5, lane = threadIdx.x & 31;
    int lrow = lane & 15, hi = lane >> 4;
    int sb = tile * 16;
    int e = texp[tile];
    int ccol = blockIdx.y * 128 + wv * 16 + lrow;
    const __bf16* arh = hhi + (size_t)(sb + lrow) * Hdim;
    const __bf16* arl = hlo + (size_t)(sb + lrow) * Hdim;
    const __bf16* brh = w2hi + ((size_t)e * Cdim + ccol) * Hdim;
    const __bf16* brl = w2lo + ((size_t)e * Cdim + ccol) * Hdim;
    int hoff = hi ? 8 : 0, hoff16 = hi ? 16 : 0;
    v8f acc = {};
    for (int kb = 0; kb < Hdim; kb += 32) {
        v16bf ah, al, bh, bl;
        loadA_pre(arh, arl, kb, hoff, ah, al);
        loadB_pre(brh, brl, kb, hoff16, bh, bl);
        WMMA(acc, ah, bh); WMMA(acc, ah, bl); WMMA(acc, al, bh);
    }
#pragma unroll
    for (int r2 = 0; r2 < 8; ++r2) {
        int row = r2 + (hi ? 8 : 0);
        slotout[(size_t)(sb + row) * Cdim + ccol] = acc[r2];
    }
}

// ---------------- Kernel 5: shared up (xs @ k_w^T + k_b), relu^2 ----------------
__global__ __launch_bounds__(256) void k_sharedup(
    const __bf16* __restrict__ xshi, const __bf16* __restrict__ xslo,
    const __bf16* __restrict__ kwhi, const __bf16* __restrict__ kwlo,
    const float* __restrict__ kb_,
    __bf16* __restrict__ kkhi, __bf16* __restrict__ kklo)
{
    int tb = blockIdx.x;
    int wv = threadIdx.x >> 5, lane = threadIdx.x & 31;
    int lrow = lane & 15, hi = lane >> 4;
    int hcol = blockIdx.y * 128 + wv * 16 + lrow;
    const __bf16* arh = xshi + (size_t)(tb * 16 + lrow) * Cdim;
    const __bf16* arl = xslo + (size_t)(tb * 16 + lrow) * Cdim;
    const __bf16* brh = kwhi + (size_t)hcol * Cdim;
    const __bf16* brl = kwlo + (size_t)hcol * Cdim;
    float bias = kb_[hcol];
    int hoff = hi ? 8 : 0, hoff16 = hi ? 16 : 0;
    v8f acc = {};
    for (int kb2 = 0; kb2 < Cdim; kb2 += 32) {
        v16bf ah, al, bh, bl;
        loadA_pre(arh, arl, kb2, hoff, ah, al);
        loadB_pre(brh, brl, kb2, hoff16, bh, bl);
        WMMA(acc, ah, bh); WMMA(acc, ah, bl); WMMA(acc, al, bh);
    }
#pragma unroll
    for (int r2 = 0; r2 < 8; ++r2) {
        int row = r2 + (hi ? 8 : 0);
        float v = fmaxf(acc[r2] + bias, 0.f);
        v = v * v;
        __bf16 h, l; splitTo(v, h, l);
        size_t o = (size_t)(tb * 16 + row) * Hdim + hcol;
        kkhi[o] = h; kklo[o] = l;
    }
}

// ---------------- Kernel 6: shared down (kk @ v_w^T + v_b) + final combine ----------------
__global__ __launch_bounds__(256) void k_shareddown(
    const __bf16* __restrict__ kkhi, const __bf16* __restrict__ kklo,
    const __bf16* __restrict__ vwhi, const __bf16* __restrict__ vwlo,
    const float* __restrict__ vb,
    const float* __restrict__ slotout, const int* __restrict__ t2s,
    const float* __restrict__ sgv, float* __restrict__ out)
{
    int tb = blockIdx.x;
    int wv = threadIdx.x >> 5, lane = threadIdx.x & 31;
    int lrow = lane & 15, hi = lane >> 4;
    int ccol = blockIdx.y * 128 + wv * 16 + lrow;
    const __bf16* arh = kkhi + (size_t)(tb * 16 + lrow) * Hdim;
    const __bf16* arl = kklo + (size_t)(tb * 16 + lrow) * Hdim;
    const __bf16* brh = vwhi + (size_t)ccol * Hdim;
    const __bf16* brl = vwlo + (size_t)ccol * Hdim;
    int hoff = hi ? 8 : 0, hoff16 = hi ? 16 : 0;
    v8f acc = {};
    for (int kb = 0; kb < Hdim; kb += 32) {
        v16bf ah, al, bh, bl;
        loadA_pre(arh, arl, kb, hoff, ah, al);
        loadB_pre(brh, brl, kb, hoff16, bh, bl);
        WMMA(acc, ah, bh); WMMA(acc, ah, bl); WMMA(acc, al, bh);
    }
#pragma unroll
    for (int r2 = 0; r2 < 8; ++r2) {
        int row = r2 + (hi ? 8 : 0);
        int t = tb * 16 + row;
        float sh = acc[r2] + vb[ccol];
        int s0 = t2s[t * 2 + 0], s1 = t2s[t * 2 + 1];
        float o = slotout[(size_t)s0 * Cdim + ccol]
                + slotout[(size_t)s1 * Cdim + ccol]
                + sgv[t] * sh;
        out[(size_t)t * Cdim + ccol] = o;
    }
}

// ---------------- Kernel 7: aux loss (deterministic reduction) ----------------
__global__ __launch_bounds__(256) void k_aux(
    const float* __restrict__ scores, const int* __restrict__ cnt,
    float* __restrict__ out)
{
    __shared__ float ssum[8];
    int tid = threadIdx.x, lane = tid & 31, wv = tid >> 5;
    float s = 0.f;
    for (int t = lane; t < Ntok; t += 32) s += scores[(size_t)t * 8 + wv];
    for (int o = 16; o > 0; o >>= 1) s += __shfl_down(s, o, 32);
    if (lane == 0) ssum[wv] = s;
    __syncthreads();
    if (tid == 0) {
        float a = 0.f;
        for (int e = 0; e < 8; ++e) {
            float mo = 0.5f * (float)cnt[e] / (float)Ntok;   // mean(onehot)
            float ms = ssum[e] / (float)Ntok;                // mean(scores)
            a += mo * ms;
        }
        out[(size_t)Ntok * Cdim] = (float)Eexp * a * 0.01f;
    }
}

extern "C" void kernel_launch(void* const* d_in, const int* in_sizes, int n_in,
                              void* d_out, int out_size, void* d_ws, size_t ws_size,
                              hipStream_t stream) {
    const float* x   = (const float*)d_in[0];
    const float* gw  = (const float*)d_in[1];
    const float* w1  = (const float*)d_in[2];
    const float* w2  = (const float*)d_in[3];
    const float* sgw = (const float*)d_in[4];
    const float* kw  = (const float*)d_in[5];
    const float* kb  = (const float*)d_in[6];
    const float* vw  = (const float*)d_in[7];
    const float* vb  = (const float*)d_in[8];
    float* out = (float*)d_out;

    char* w = (char*)d_ws;
    size_t o = 0;
    auto alloc = [&](size_t bytes) -> char* {
        char* p = w + o;
        o = (o + bytes + 255) & ~(size_t)255;
        return p;
    };
    const size_t NW1 = (size_t)Eexp * Hdim * Cdim;   // 16.78M
    const size_t NW2 = (size_t)Eexp * Cdim * Hdim;   // 16.78M
    const size_t NKW = (size_t)Hdim * Cdim;          // 2.10M
    const size_t NVW = (size_t)Cdim * Hdim;          // 2.10M

    __bf16* xthi   = (__bf16*)alloc((size_t)Ntok * Cdim * 2);
    __bf16* xtlo   = (__bf16*)alloc((size_t)Ntok * Cdim * 2);
    __bf16* xshi   = (__bf16*)alloc((size_t)Ntok * Cdim * 2);
    __bf16* xslo   = (__bf16*)alloc((size_t)Ntok * Cdim * 2);
    __bf16* w1hi   = (__bf16*)alloc(NW1 * 2);
    __bf16* w1lo   = (__bf16*)alloc(NW1 * 2);
    __bf16* w2hi   = (__bf16*)alloc(NW2 * 2);
    __bf16* w2lo   = (__bf16*)alloc(NW2 * 2);
    __bf16* kwhi   = (__bf16*)alloc(NKW * 2);
    __bf16* kwlo   = (__bf16*)alloc(NKW * 2);
    __bf16* vwhi   = (__bf16*)alloc(NVW * 2);
    __bf16* vwlo   = (__bf16*)alloc(NVW * 2);
    __bf16* hhi    = (__bf16*)alloc((size_t)SCAP * Hdim * 2);
    __bf16* hlo    = (__bf16*)alloc((size_t)SCAP * Hdim * 2);
    __bf16* kkhi   = (__bf16*)alloc((size_t)Ntok * Hdim * 2);
    __bf16* kklo   = (__bf16*)alloc((size_t)Ntok * Hdim * 2);
    float* slotout = (float*)alloc((size_t)SCAP * Cdim * 4);
    float* scores  = (float*)alloc((size_t)Ntok * Eexp * 4);
    float* tkw     = (float*)alloc((size_t)Ntok * 2 * 4);
    float* sgv     = (float*)alloc((size_t)Ntok * 4);
    float* pw      = (float*)alloc((size_t)SCAP * 4);
    int*   tidx    = (int*)alloc((size_t)Ntok * 2 * 4);
    int*   cnt     = (int*)alloc((size_t)Eexp * 4);
    int*   off     = (int*)alloc((size_t)(Eexp + 1) * 4);
    int*   perm    = (int*)alloc((size_t)SCAP * 4);
    int*   t2s     = (int*)alloc((size_t)Ntok * 2 * 4);
    int*   texp    = (int*)alloc((size_t)RT * 4);
    // total ws usage ~235 MB

    // one-time operand pre-split (bandwidth-bound, ~13us)
    k_split<<<(int)(NW1 / 4 + 255) / 256, 256, 0, stream>>>(w1, w1hi, w1lo, (int)(NW1 / 4));
    k_split<<<(int)(NW2 / 4 + 255) / 256, 256, 0, stream>>>(w2, w2hi, w2lo, (int)(NW2 / 4));
    k_split<<<(int)(NKW / 4 + 255) / 256, 256, 0, stream>>>(kw, kwhi, kwlo, (int)(NKW / 4));
    k_split<<<(int)(NVW / 4 + 255) / 256, 256, 0, stream>>>(vw, vwhi, vwlo, (int)(NVW / 4));

    k_gating<<<Ntok, 256, 0, stream>>>(x, gw, sgw, xthi, xtlo, xshi, xslo,
                                       scores, tidx, tkw, sgv);
    k_scatter<<<1, 256, 0, stream>>>(tidx, tkw, cnt, off, perm, pw, t2s, texp);
    k_up<<<dim3(RT, Hdim / 128), 256, 0, stream>>>(xthi, xtlo, w1hi, w1lo,
                                                   perm, pw, texp, hhi, hlo);
    k_down<<<dim3(RT, Cdim / 128), 256, 0, stream>>>(hhi, hlo, w2hi, w2lo, texp, slotout);
    k_sharedup<<<dim3(Ntok / 16, Hdim / 128), 256, 0, stream>>>(xshi, xslo, kwhi, kwlo,
                                                                kb, kkhi, kklo);
    k_shareddown<<<dim3(Ntok / 16, Cdim / 128), 256, 0, stream>>>(kkhi, kklo, vwhi, vwlo,
                                                                  vb, slotout, t2s, sgv, out);
    k_aux<<<1, 256, 0, stream>>>(scores, cnt, out);
}
